// MultiheadAttention_58763742544160
// MI455X (gfx1250) — matrix operand order repaired and checked by reading, not compiled, not run on hardware
//
#include <hip/hip_runtime.h>
#include <hip/hip_bf16.h>
#include <math.h>

// MultiheadAttention forward for MI455X (gfx1250, wave32, WMMA).
// Pipeline: qkv_proj (WMMA GEMM, fused head-split + bf16 cvt)
//        -> attn (flash-style online softmax, all-WMMA, transposed tiles)
//        -> out_proj (WMMA GEMM, async-to-LDS A-tiles, f32 epilogue).

#define T_DIM 2048
#define B_DIM 2
#define E_DIM 1024
#define H_DIM 16
#define DH    64
#define MROWS (T_DIM * B_DIM)   // 4096

typedef __bf16 bf16_t;
typedef __attribute__((ext_vector_type(8)))  __bf16 v8bf;
typedef __attribute__((ext_vector_type(16))) __bf16 v16bf;
typedef __attribute__((ext_vector_type(8)))  float  v8f;
typedef __attribute__((ext_vector_type(4)))  float  v4f;
typedef __attribute__((ext_vector_type(4)))  int    v4i;

#if defined(__has_builtin)
#if __has_builtin(__builtin_amdgcn_global_load_async_to_lds_b128)
#define HAS_ASYNC_LDS 1
#endif
#endif
#ifndef HAS_ASYNC_LDS
#define HAS_ASYNC_LDS 0
#endif

#define GLOBAL_AS __attribute__((address_space(1)))
#define LDS_AS    __attribute__((address_space(3)))

static __device__ inline void wait_async_lds() {
#if HAS_ASYNC_LDS
  asm volatile("s_wait_asynccnt 0" ::: "memory");
#endif
}

static __device__ inline v16bf join16(v8bf lo, v8bf hi) {
  return __builtin_shufflevector(lo, hi, 0,1,2,3,4,5,6,7,8,9,10,11,12,13,14,15);
}

// A-matrix fragment (16x32 bf16, MxK). rowPtr = &src[row][k_window_base].
// a[j] = A[m][16*h + j]  -> one contiguous 32B run.
static __device__ inline v16bf load_a_frag(const bf16_t* rowPtr, int h) {
  v8bf lo = *(const v8bf*)(rowPtr + 16 * h);
  v8bf hi = *(const v8bf*)(rowPtr + 16 * h + 8);
  return join16(lo, hi);
}

// B-matrix fragment (32x16 bf16, KxN) from an n-major (k-contiguous) source.
// b[j] = B[16*h + j][n]  -> one contiguous 32B run.
static __device__ inline v16bf load_b_frag(const bf16_t* rowPtr, int h) {
  v8bf lo = *(const v8bf*)(rowPtr + 16 * h);
  v8bf hi = *(const v8bf*)(rowPtr + 16 * h + 8);
  return join16(lo, hi);
}

static __device__ inline v8f wmma_bf16(v16bf a, v16bf b, v8f c) {
  return __builtin_amdgcn_wmma_f32_16x16x32_bf16(false, a, false, b,
                                                 (short)0, c, false, false);
}

// Stage 16 contiguous elements into LDS (f32 source: convert to bf16).
static __device__ inline void stage_tile(const float* src, bf16_t* dst) {
  v4f a = *(const v4f*)(src);
  v4f b = *(const v4f*)(src + 4);
  v4f c = *(const v4f*)(src + 8);
  v4f d = *(const v4f*)(src + 12);
  v8bf t0, t1;
  t0[0] = (bf16_t)a[0]; t0[1] = (bf16_t)a[1];
  t0[2] = (bf16_t)a[2]; t0[3] = (bf16_t)a[3];
  t0[4] = (bf16_t)b[0]; t0[5] = (bf16_t)b[1];
  t0[6] = (bf16_t)b[2]; t0[7] = (bf16_t)b[3];
  t1[0] = (bf16_t)c[0]; t1[1] = (bf16_t)c[1];
  t1[2] = (bf16_t)c[2]; t1[3] = (bf16_t)c[3];
  t1[4] = (bf16_t)d[0]; t1[5] = (bf16_t)d[1];
  t1[6] = (bf16_t)d[2]; t1[7] = (bf16_t)d[3];
  *(v8bf*)dst = t0;
  *(v8bf*)(dst + 8) = t1;
}

// Stage 16 contiguous bf16 (32 bytes): raw copy -> use async global->LDS DMA
// (no VGPR round trip, ASYNCcnt-tracked) when the toolchain exposes it.
static __device__ inline void stage_tile(const bf16_t* src, bf16_t* dst) {
#if HAS_ASYNC_LDS
  __builtin_amdgcn_global_load_async_to_lds_b128(
      (GLOBAL_AS v4i*)src, (LDS_AS v4i*)dst, 0, 0);
  __builtin_amdgcn_global_load_async_to_lds_b128(
      (GLOBAL_AS v4i*)src, (LDS_AS v4i*)dst, 16, 0);
#else
  *(v8bf*)dst = *(const v8bf*)src;
  *(v8bf*)(dst + 8) = *(const v8bf*)(src + 8);
#endif
}

// ---------------------------------------------------------------------------
// Unified GEMM: C[m][n] = sum_k X[m][k] * W[n][k] + bias[n]
// Block 256 thr (8 waves, 4x2), tile 64(M) x 64(N), K-step 64,
// double-buffered LDS (one barrier + 4 WMMA per wave per K-step).
// OUTMODE 0: scatter bf16 to per-head layout (QKV projection, scaled).
// OUTMODE 1: f32 row-major output (out projection).
// ---------------------------------------------------------------------------
template <typename TIn, int OUTMODE>
__global__ __launch_bounds__(256) void gemm_kernel(
    const TIn* __restrict__ X, const float* __restrict__ W,
    const float* __restrict__ bias, void* __restrict__ outPtr,
    int transposedV, float scale) {
  __shared__ bf16_t As[2][64][64];   // [buf][m][k]
  __shared__ bf16_t Bs[2][64][64];   // [buf][n][k]  (n-major, k contiguous)
  const int tid = threadIdx.x;
  const int lane = tid & 31, h = lane >> 4, ln = lane & 15;
  const int w = tid >> 5, wm = w >> 1, wn = w & 1;   // 4x2 wave grid
  const int m0 = blockIdx.y * 64;
  const int n0 = blockIdx.x * 64;

  const int srow = tid >> 2;           // 0..63
  const int scol = (tid & 3) * 16;     // 0,16,32,48
  const TIn*   xsrc = X + (size_t)(m0 + srow) * E_DIM + scol;
  const float* wsrc = W + (size_t)(n0 + srow) * E_DIM + scol;

  v8f acc0 = {}, acc1 = {};

  // prologue: stage k-tile 0 into buffer 0
  stage_tile(xsrc, &As[0][srow][scol]);
  stage_tile(wsrc, &Bs[0][srow][scol]);
  wait_async_lds();
  __syncthreads();

  const int KT = E_DIM / 64;   // 16
  for (int kt = 0; kt < KT; ++kt) {
    const int buf = kt & 1;
    if (kt + 1 < KT) {   // issue next tile's loads before computing this one
      stage_tile(xsrc + (kt + 1) * 64, &As[buf ^ 1][srow][scol]);
      stage_tile(wsrc + (kt + 1) * 64, &Bs[buf ^ 1][srow][scol]);
    }
    const bf16_t* arow  = &As[buf][wm * 16 + ln][0];
    const bf16_t* brow0 = &Bs[buf][wn * 32 + ln][0];
    const bf16_t* brow1 = &Bs[buf][wn * 32 + 16 + ln][0];
    v16bf af0 = load_a_frag(arow, h);
    v16bf af1 = load_a_frag(arow + 32, h);
    acc0 = wmma_bf16(af0, load_b_frag(brow0, h), acc0);
    acc0 = wmma_bf16(af1, load_b_frag(brow0 + 32, h), acc0);
    acc1 = wmma_bf16(af0, load_b_frag(brow1, h), acc1);
    acc1 = wmma_bf16(af1, load_b_frag(brow1 + 32, h), acc1);
    wait_async_lds();
    __syncthreads();
  }

  #pragma unroll
  for (int j = 0; j < 2; ++j) {
    v8f acc = j ? acc1 : acc0;
    const int e  = n0 + wn * 32 + j * 16 + ln;
    const float be = bias[e];
    if constexpr (OUTMODE == 0) {
      bf16_t* Out = (bf16_t*)outPtr;
      const int hh = e >> 6, d = e & 63;
      #pragma unroll
      for (int r = 0; r < 8; ++r) {
        const int m = m0 + wm * 16 + r + 8 * h;
        const int t = m >> 1, b = m & 1;           // m = t*B + b (B=2)
        const int bh = b * H_DIM + hh;
        const float v = (acc[r] + be) * scale;
        const size_t idx = transposedV
            ? ((size_t)bh * DH + d) * T_DIM + t
            : ((size_t)bh * T_DIM + t) * DH + d;
        Out[idx] = (bf16_t)v;
      }
    } else {
      float* Out = (float*)outPtr;
      #pragma unroll
      for (int r = 0; r < 8; ++r) {
        const int m = m0 + wm * 16 + r + 8 * h;
        Out[(size_t)m * E_DIM + e] = acc[r] + be;
      }
    }
  }
}

// ---------------------------------------------------------------------------
// Attention: per wave 16 queries, full Dh=64, streams keys in chunks of 64.
// Computes S^T = K * Q^T and ctx^T = V^T * P so each lane owns one query
// column -> online softmax is in-lane reduction + one shfl_xor(16).
// K/V (8 MB each) are L2-resident (192 MB L2) -> fragments loaded straight
// from global, no LDS, no barriers.
// ---------------------------------------------------------------------------
__global__ __launch_bounds__(128) void attn_kernel(
    const bf16_t* __restrict__ Q,   // (32, 2048, 64), pre-scaled by log2e/8
    const bf16_t* __restrict__ K,   // (32, 2048, 64)
    const bf16_t* __restrict__ Vt,  // (32, 64, 2048)
    bf16_t* __restrict__ Ctx)       // (4096, 1024) token-major
{
  const int lane = threadIdx.x & 31, h = lane >> 4, ln = lane & 15;
  const int w = threadIdx.x >> 5;
  const int tq = blockIdx.x * 64 + w * 16 + ln;   // this lane's query column
  const int bh = blockIdx.y;

  const bf16_t* qrow = Q + ((size_t)bh * T_DIM + tq) * DH;
  const v16bf qf0 = load_b_frag(qrow, h);        // d in [0,32)
  const v16bf qf1 = load_b_frag(qrow + 32, h);   // d in [32,64)

  const bf16_t* Kbase = K  + (size_t)bh * T_DIM * DH;
  const bf16_t* Vbase = Vt + (size_t)bh * DH * T_DIM;

  v8f ctx[4] = {};                 // ctx^T tiles: rows d = 16i + r + 8h
  float m_run = -1e30f, l_run = 0.f;

  for (int s0 = 0; s0 < T_DIM; s0 += 64) {
    // S^T chunk: 64 keys x 16 queries, 4 C-tiles stacked along keys
    v8f st[4];
    #pragma unroll
    for (int i = 0; i < 4; ++i) {
      const bf16_t* krow = Kbase + (size_t)(s0 + 16 * i + ln) * DH;
      v16bf ka0 = load_a_frag(krow, h);
      v16bf ka1 = load_a_frag(krow + 32, h);
      v8f c = {};
      c = wmma_bf16(ka0, qf0, c);
      c = wmma_bf16(ka1, qf1, c);
      st[i] = c;
    }
    // online softmax (scores already in log2 units)
    float mc = -1e30f;
    #pragma unroll
    for (int i = 0; i < 4; ++i)
      #pragma unroll
      for (int r = 0; r < 8; ++r) mc = fmaxf(mc, st[i][r]);
    mc = fmaxf(mc, __shfl_xor(mc, 16, 32));
    const float m_new = fmaxf(m_run, mc);
    const float alpha = exp2f(m_run - m_new);
    float p[4][8];
    float rs = 0.f;
    #pragma unroll
    for (int i = 0; i < 4; ++i)
      #pragma unroll
      for (int r = 0; r < 8; ++r) {
        p[i][r] = exp2f(st[i][r] - m_new);
        rs += p[i][r];
      }
    rs += __shfl_xor(rs, 16, 32);
    l_run = l_run * alpha + rs;
    m_run = m_new;
    #pragma unroll
    for (int i = 0; i < 4; ++i)
      #pragma unroll
      for (int r = 0; r < 8; ++r) ctx[i][r] *= alpha;

    // Build P B-fragments (keys are the K-dim): fix half-wave packing with
    // shfl_xor(16); b[j] = P[key = 32c + 16h + j][q = ln].
    v16bf pf[2];
    #pragma unroll
    for (int c = 0; c < 2; ++c) {
      v16bf f = {};
      #pragma unroll
      for (int j = 0; j < 8; ++j) {
        const float o0 = __shfl_xor(p[2 * c + 1][j], 16, 32);
        const float o1 = __shfl_xor(p[2 * c][j],     16, 32);
        f[j]     = (bf16_t)((h == 0) ? p[2 * c][j]     : o0);
        f[j + 8] = (bf16_t)((h == 1) ? p[2 * c + 1][j] : o1);
      }
      pf[c] = f;
    }
    // ctx^T += V^T * P
    #pragma unroll
    for (int i = 0; i < 4; ++i) {
      const bf16_t* vrow = Vbase + (size_t)(16 * i + ln) * T_DIM + s0;
      v16bf va0 = load_a_frag(vrow, h);
      v16bf va1 = load_a_frag(vrow + 32, h);
      ctx[i] = wmma_bf16(va0, pf[0], ctx[i]);
      ctx[i] = wmma_bf16(va1, pf[1], ctx[i]);
    }
  }

  const float inv = 1.0f / l_run;
  const int b = bh >> 4, hh = bh & 15;
  #pragma unroll
  for (int i = 0; i < 4; ++i)
    #pragma unroll
    for (int r = 0; r < 8; ++r) {
      const int d = 16 * i + r + 8 * h;
      const size_t idx = ((size_t)tq * B_DIM + b) * E_DIM + hh * DH + d;
      Ctx[idx] = (bf16_t)(ctx[i][r] * inv);
    }
}

// ---------------------------------------------------------------------------
extern "C" void kernel_launch(void* const* d_in, const int* in_sizes, int n_in,
                              void* d_out, int out_size, void* d_ws,
                              size_t ws_size, hipStream_t stream) {
  (void)in_sizes; (void)n_in; (void)out_size; (void)ws_size;
  const float* query = (const float*)d_in[0];
  const float* key   = (const float*)d_in[1];
  const float* value = (const float*)d_in[2];
  const float* Win   = (const float*)d_in[3];   // (3E, E)
  const float* bin   = (const float*)d_in[4];   // (3E)
  const float* Wout  = (const float*)d_in[5];   // (E, E)
  const float* bout  = (const float*)d_in[6];   // (E)
  float* out = (float*)d_out;

  char* ws = (char*)d_ws;
  const size_t SEG = (size_t)8 * 1024 * 1024;   // 8 MB per bf16 tensor
  bf16_t* Q   = (bf16_t*)(ws);
  bf16_t* Kk  = (bf16_t*)(ws + SEG);
  bf16_t* Vt  = (bf16_t*)(ws + 2 * SEG);
  bf16_t* Ctx = (bf16_t*)(ws + 3 * SEG);

  const dim3 gblk(E_DIM / 64, MROWS / 64);      // (16, 64)
  // fold 1/sqrt(Dh) * log2(e) into Q so softmax uses exp2 directly
  const float qscale = 0.125f * 1.44269504088896340736f;

  gemm_kernel<float, 0><<<gblk, 256, 0, stream>>>(query, Win, bin, Q, 0,
                                                  qscale);
  gemm_kernel<float, 0><<<gblk, 256, 0, stream>>>(
      key, Win + (size_t)E_DIM * E_DIM, bin + E_DIM, Kk, 0, 1.0f);
  gemm_kernel<float, 0><<<gblk, 256, 0, stream>>>(
      value, Win + (size_t)2 * E_DIM * E_DIM, bin + 2 * E_DIM, Vt, 1, 1.0f);
  attn_kernel<<<dim3(T_DIM / 64, B_DIM * H_DIM), 128, 0, stream>>>(Q, Kk, Vt,
                                                                   Ctx);
  gemm_kernel<bf16_t, 1><<<gblk, 256, 0, stream>>>(Ctx, Wout, bout, out, 0,
                                                   1.0f);
}
